// MemoryReader_15126874816724
// MI455X (gfx1250) — compile-verified, matrix operand order checked by reading, not run on hardware
//
#include <hip/hip_runtime.h>
#include <math.h>

typedef _Float16 v16h __attribute__((ext_vector_type(16)));
typedef _Float16 v8h  __attribute__((ext_vector_type(8)));
typedef float    v8f  __attribute__((ext_vector_type(8)));
typedef uint32_t u32x4 __attribute__((ext_vector_type(4)));
typedef int      i32x4 __attribute__((ext_vector_type(4)));
typedef int      i32x8 __attribute__((ext_vector_type(8)));

namespace {
constexpr int kB   = 4;
constexpr int kCK  = 64;
constexpr int kCV  = 512;
constexpr int kTHW = 7200;   // 8*30*30
constexpr int kHW  = 900;    // 30*30
constexpr int kQS  = 32;     // queries per workgroup (2 WMMA n-tiles)
constexpr int kMC  = 32;     // memory chunk = K of value GEMM
constexpr int kNCH = kTHW / kMC;  // 225, exact
constexpr int kST  = 72;     // LDS row stride in halves (16B-aligned, bank-padded)

union Frag {
  v16h v;
  v8h  h[2];
  _Float16 e[16];
};
} // namespace

#if __has_builtin(__builtin_amdgcn_tensor_load_to_lds)
#define MR_HAVE_TDM 1
#else
#define MR_HAVE_TDM 0
#endif

#if MR_HAVE_TDM
// Issue one TDM DMA: 2-D tile (tile_dim0 = 32 contiguous m, tile_dim1 = 64 rows of
// stride THW) of f32 from global into LDS at byte offset lds_off ([c][m] order).
__device__ __forceinline__ void tdm_load_mk_tile(const float* gsrc, uint32_t lds_off)
{
  const uint64_t ga = (uint64_t)(uintptr_t)gsrc;
  u32x4 g0;
  g0[0] = 1u;                                                  // count=1 (valid user D#)
  g0[1] = lds_off;                                             // LDS dest byte address
  g0[2] = (uint32_t)ga;                                        // global_addr[31:0]
  g0[3] = (uint32_t)((ga >> 32) & 0x01FFFFFFu) | (2u << 30);   // global_addr[56:32] | type=2
  i32x8 g1;
  g1[0] = (int)(2u << 16);                                     // data_size = 4 bytes
  g1[1] = (int)(((uint32_t)kTHW & 0xFFFFu) << 16);             // tensor_dim0[15:0]
  g1[2] = (int)(((uint32_t)kTHW >> 16) | ((uint32_t)kCK << 16)); // dim0[31:16] | dim1[15:0]
  g1[3] = (int)((uint32_t)kMC << 16);                          // dim1[31:16]=0 | tile_dim0=32
  g1[4] = (int)(uint32_t)kCK;                                  // tile_dim1=64, tile_dim2=0
  g1[5] = (int)(uint32_t)kTHW;                                 // tensor_dim0_stride[31:0]
  g1[6] = 0;                                                   // stride0[47:32] | stride1[15:0]
  g1[7] = 0;                                                   // stride1[47:16]
  const i32x4 gz = {0, 0, 0, 0};
#if defined(__clang_major__) && (__clang_major__ >= 23)
  const i32x8 gz8 = {0, 0, 0, 0, 0, 0, 0, 0};
  __builtin_amdgcn_tensor_load_to_lds(g0, g1, gz, gz, gz8, 0);
#else
  __builtin_amdgcn_tensor_load_to_lds(g0, g1, gz, gz, 0);
#endif
}
#endif

__global__ __launch_bounds__(256, 2)
void mr_attn_kernel(const float* __restrict__ mk,
                    const float* __restrict__ qk,
                    const float* __restrict__ mv,
                    float* __restrict__ out)
{
  __shared__ __attribute__((aligned(16))) _Float16 lds_qk[kQS * kST];
  __shared__ __attribute__((aligned(16))) _Float16 lds_mk[kMC * kST];
  __shared__ float lds_asq[kMC];
#if MR_HAVE_TDM
  __shared__ __attribute__((aligned(16))) float lds_raw[2][kCK * kMC]; // TDM dst, [c][m]
#endif

  const int tid  = threadIdx.x;
  const int lane = tid & 31;
  const int wv   = tid >> 5;       // wave 0..7 (wave32)
  const int ln   = lane & 15;      // column / row-within-tile
  const int hi   = lane >> 4;      // lane half
  const int mo   = hi << 3;        // K / M offset for upper lane half

  const int q0  = blockIdx.x * kQS;
  const int cvh = blockIdx.y;      // 0..1 : which half of CV
  const int b   = blockIdx.z;

  const float* mkB = mk + (size_t)b * kCK * kTHW;
  const float* qkB = qk + (size_t)b * kCK * kHW;
  const float* mvB = mv + (size_t)b * kCV * kTHW;

#if MR_HAVE_TDM
  const uint32_t raw_off0 = (uint32_t)(uintptr_t)&lds_raw[0][0]; // low 32 bits = LDS offset
  const uint32_t raw_off1 = (uint32_t)(uintptr_t)&lds_raw[1][0];
  if (wv == 0) tdm_load_mk_tile(mkB, raw_off0);                  // prime chunk 0
#endif

  // ---- stage qk[64c x 32q] -> lds_qk[q][c] as f16 (zero-pad q >= 900) ----
  for (int e = tid; e < kCK * kQS; e += 256) {
    const int c = e >> 5, j = e & 31, q = q0 + j;
    const float v = (q < kHW) ? qkB[(size_t)c * kHW + q] : 0.f;
    lds_qk[j * kST + c] = (_Float16)v;
  }
  __syncthreads();

  // ---- qk B-fragments [khalf][qsub], loop-invariant ----
  Frag qkf[2][2];
#pragma unroll
  for (int h = 0; h < 2; ++h)
#pragma unroll
    for (int s = 0; s < 2; ++s) {
      const int j = s * 16 + ln, bc = h * 32 + mo;
      qkf[h][s].h[0] = *(const v8h*)&lds_qk[j * kST + bc];
      qkf[h][s].h[1] = *(const v8h*)&lds_qk[j * kST + bc + 16];
    }

  v8f acc[2][2];
#pragma unroll
  for (int t = 0; t < 2; ++t)
#pragma unroll
    for (int s = 0; s < 2; ++s)
      acc[t][s] = (v8f){0.f,0.f,0.f,0.f,0.f,0.f,0.f,0.f};

  float rmax[2] = {-1e30f, -1e30f};
  float rsum[2] = {0.f, 0.f};

  const int cvb = cvh * 256 + wv * 32;   // this wave's 32 CV rows

  for (int ch = 0; ch < kNCH; ++ch) {
    const int m0 = ch * kMC;

#if MR_HAVE_TDM
    if (wv == 0) __builtin_amdgcn_s_wait_tensorcnt(0);  // chunk ch landed in LDS
    __syncthreads();            // publish tile + previous chunk's LDS reads done
    if (wv == 0 && ch + 1 < kNCH)                       // DMA next chunk, other buffer
      tdm_load_mk_tile(mkB + (size_t)(ch + 1) * kMC, ((ch + 1) & 1) ? raw_off1 : raw_off0);
    // convert TDM tile ([c][m] f32) -> lds_mk ([m][c] f16)
    {
      const float* raw = &lds_raw[ch & 1][0];
      for (int e = tid; e < kCK * kMC; e += 256) {
        const int c = e >> 5, mm = e & 31;
        lds_mk[mm * kST + c] = (_Float16)raw[c * kMC + mm];
      }
    }
#else
    __syncthreads();            // previous chunk's LDS reads done
    for (int e = tid; e < kCK * kMC; e += 256) {
      const int c = e >> 5, mm = e & 31;
      lds_mk[mm * kST + c] = (_Float16)mkB[(size_t)c * kTHW + m0 + mm];
    }
#endif
    __syncthreads();

    // ---- |mk|^2 per m from the staged tile ----
    if (tid < kMC) {
      float ss = 0.f;
#pragma unroll
      for (int k = 0; k < 8; ++k) {
        const v8h hv = *(const v8h*)&lds_mk[tid * kST + k * 8];
#pragma unroll
        for (int j = 0; j < 8; ++j) { const float f = (float)hv[j]; ss += f * f; }
      }
      lds_asq[tid] = ss;
    }
    __syncthreads();

    // ---- mk A-fragments [msub][khalf] ----
    Frag mkf[2][2];
#pragma unroll
    for (int ms = 0; ms < 2; ++ms)
#pragma unroll
      for (int h = 0; h < 2; ++h) {
        const int row = ms * 16 + ln, bc = h * 32 + mo;
        mkf[ms][h].h[0] = *(const v8h*)&lds_mk[row * kST + bc];
        mkf[ms][h].h[1] = *(const v8h*)&lds_mk[row * kST + bc + 16];
      }

    // ---- mv A-fragments straight from global (f32 -> f16) + prefetch ----
    Frag mvf[2];
#pragma unroll
    for (int t = 0; t < 2; ++t) {
      const float* p = mvB + (size_t)(cvb + t * 16 + ln) * kTHW + m0 + mo;
      const float4 f0 = *(const float4*)(p);
      const float4 f1 = *(const float4*)(p + 4);
      const float4 f2 = *(const float4*)(p + 16);
      const float4 f3 = *(const float4*)(p + 20);
      mvf[t].e[0]=(_Float16)f0.x; mvf[t].e[1]=(_Float16)f0.y;
      mvf[t].e[2]=(_Float16)f0.z; mvf[t].e[3]=(_Float16)f0.w;
      mvf[t].e[4]=(_Float16)f1.x; mvf[t].e[5]=(_Float16)f1.y;
      mvf[t].e[6]=(_Float16)f1.z; mvf[t].e[7]=(_Float16)f1.w;
      mvf[t].e[8]=(_Float16)f2.x; mvf[t].e[9]=(_Float16)f2.y;
      mvf[t].e[10]=(_Float16)f2.z; mvf[t].e[11]=(_Float16)f2.w;
      mvf[t].e[12]=(_Float16)f3.x; mvf[t].e[13]=(_Float16)f3.y;
      mvf[t].e[14]=(_Float16)f3.z; mvf[t].e[15]=(_Float16)f3.w;
      if (ch + 1 < kNCH) __builtin_prefetch(p + kMC, 0, 1);
    }

    float asq0[8], asq1[8];
#pragma unroll
    for (int i = 0; i < 8; ++i) {
      asq0[i] = lds_asq[i + mo];
      asq1[i] = lds_asq[16 + i + mo];
    }

    const v8f zero = (v8f){0.f,0.f,0.f,0.f,0.f,0.f,0.f,0.f};
#pragma unroll
    for (int s = 0; s < 2; ++s) {
      // logits for m-subtiles 0/1, K=64 via two WMMAs each
      v8f d0 = zero, d1 = zero;
      d0 = __builtin_amdgcn_wmma_f32_16x16x32_f16(false, mkf[0][0].v, false, qkf[0][s].v, (short)0, d0, false, false);
      d0 = __builtin_amdgcn_wmma_f32_16x16x32_f16(false, mkf[0][1].v, false, qkf[1][s].v, (short)0, d0, false, false);
      d1 = __builtin_amdgcn_wmma_f32_16x16x32_f16(false, mkf[1][0].v, false, qkf[0][s].v, (short)0, d1, false, false);
      d1 = __builtin_amdgcn_wmma_f32_16x16x32_f16(false, mkf[1][1].v, false, qkf[1][s].v, (short)0, d1, false, false);

      // (2ab - |a|^2)/sqrt(64), online softmax along m (in-lane + one xor-16)
      float lg0[8], lg1[8];
      float cmax = -1e30f;
#pragma unroll
      for (int i = 0; i < 8; ++i) {
        lg0[i] = 0.25f * d0[i] - 0.125f * asq0[i];
        lg1[i] = 0.25f * d1[i] - 0.125f * asq1[i];
        cmax = fmaxf(cmax, fmaxf(lg0[i], lg1[i]));
      }
      cmax = fmaxf(cmax, __shfl_xor(cmax, 16));
      const float nm = fmaxf(rmax[s], cmax);
      const float sc = __expf(rmax[s] - nm);
      rmax[s] = nm;

      Frag pf;                 // probability B-fragment: pure in-lane packing
      float psum = 0.f;
#pragma unroll
      for (int i = 0; i < 8; ++i) {
        const float p0 = __expf(lg0[i] - nm);
        const float p1 = __expf(lg1[i] - nm);
        psum += p0 + p1;
        pf.e[i]     = (_Float16)p0;
        pf.e[8 + i] = (_Float16)p1;
      }
      rsum[s] = rsum[s] * sc + psum;

#pragma unroll
      for (int t = 0; t < 2; ++t) {
#pragma unroll
        for (int i = 0; i < 8; ++i) acc[t][s][i] *= sc;   // per-lane column rescale
        acc[t][s] = __builtin_amdgcn_wmma_f32_16x16x32_f16(
            false, mvf[t].v, false, pf.v, (short)0, acc[t][s], false, false);
      }
    }
  }

  // ---- epilogue: normalize and store mem part of the output ----
#pragma unroll
  for (int s = 0; s < 2; ++s) {
    const float tot = rsum[s] + __shfl_xor(rsum[s], 16);
    const float inv = (tot > 0.f) ? (1.f / tot) : 0.f;
    const int q = q0 + s * 16 + ln;
    if (q < kHW) {
#pragma unroll
      for (int t = 0; t < 2; ++t)
#pragma unroll
        for (int i = 0; i < 8; ++i)
          out[(size_t)b * (2 * kCV * kHW) +
              (size_t)(cvb + t * 16 + mo + i) * kHW + q] = acc[t][s][i] * inv;
    }
  }
}

// Copy qv into the second CV channels of the output: out4[i + (b+1)*115200] = qv4[i]
__global__ __launch_bounds__(256)
void qv_copy_kernel(const float4* __restrict__ qv4, float4* __restrict__ out4, int n4)
{
  const int i = blockIdx.x * blockDim.x + threadIdx.x;
  if (i < n4) {
    const int b = i / (kCV * kHW / 4);   // 115200 float4 per batch
    out4[i + (size_t)(b + 1) * (kCV * kHW / 4)] = qv4[i];
  }
}

extern "C" void kernel_launch(void* const* d_in, const int* in_sizes, int n_in,
                              void* d_out, int out_size, void* d_ws, size_t ws_size,
                              hipStream_t stream) {
  const float* mk = (const float*)d_in[0];
  const float* qk = (const float*)d_in[1];
  const float* mv = (const float*)d_in[2];
  const float* qv = (const float*)d_in[3];
  float* out = (float*)d_out;

  dim3 grid((kHW + kQS - 1) / kQS, 2, kB);   // 29 x 2 x 4 = 232 WGs
  mr_attn_kernel<<<grid, 256, 0, stream>>>(mk, qk, mv, out);

  const int n4 = kB * kCV * kHW / 4;         // 460800 float4
  qv_copy_kernel<<<(n4 + 255) / 256, 256, 0, stream>>>(
      (const float4*)qv, (float4*)out, n4);
}
